// ScaledDotProductAttentionWithReference2_87634512708216
// MI455X (gfx1250) — matrix operand order repaired and checked by reading, not compiled, not run on hardware
//
#include <hip/hip_runtime.h>

// ---------------------------------------------------------------------------
// Flash-attention with attention-sink column + deterministic dropout.
// MI455X (gfx1250): wave32, v_wmma_f32_16x16x32_bf16 (f32 accum),
// TDM tensor_load_to_lds double-buffered DMA for the dropout tensor `u`,
// global_prefetch for next K/V tiles.
// ---------------------------------------------------------------------------

typedef __bf16 v16bf __attribute__((ext_vector_type(16)));
typedef __bf16 v8bf  __attribute__((ext_vector_type(8)));
typedef float  v8f   __attribute__((ext_vector_type(8)));

typedef unsigned int u32x4 __attribute__((ext_vector_type(4)));
typedef int          i32x4 __attribute__((ext_vector_type(4)));
typedef int          i32x8 __attribute__((ext_vector_type(8)));

#define B_   16
#define LQ   2048
#define LK   2048
#define D_   128
#define TQ   128           // queries per block (8 waves x 16)
#define TK   64            // keys per k-iteration
#define NW   8             // waves per block
#define NT   (LK / TK)     // k-iterations

#if __has_builtin(__builtin_amdgcn_tensor_load_to_lds) && \
    __has_builtin(__builtin_amdgcn_s_wait_tensorcnt)
#define USE_TDM 1
#else
#define USE_TDM 0
#endif

__device__ __forceinline__ float warp_max_half16(float x) {
  #pragma unroll
  for (int m = 1; m <= 8; m <<= 1) x = fmaxf(x, __shfl_xor(x, m, 32));
  return x;
}
__device__ __forceinline__ float warp_sum_half16(float x) {
  #pragma unroll
  for (int m = 1; m <= 8; m <<= 1) x += __shfl_xor(x, m, 32);
  return x;
}

#if USE_TDM
// Issue one TDM DMA: copy a 16x64 f32 tile (row stride LK+1 floats) from
// global `gsrc` into LDS at `ldst`.  Descriptor per CDNA5 ISA ch.8 (D#).
// Wave-uniform arguments -> descriptor groups live in SGPRs; EXEC ignored.
// This toolchain exposes the 6-arg builtin:
//   (u32x4 g0, i32x8 g1, i32x4 g2, i32x4 g3, i32x8 g4, i32 cpol)
__device__ __forceinline__ void tdm_load_u_tile(const float* gsrc, void* ldst) {
  const unsigned long long ga = (unsigned long long)(uintptr_t)gsrc;
  const unsigned int       la = (unsigned int)(uintptr_t)ldst; // LDS byte addr

  u32x4 g0;
  g0[0] = 1u;                                        // count=1 (valid), user D#
  g0[1] = la;                                        // lds_addr[31:0]
  g0[2] = (unsigned int)ga;                          // global_addr[31:0]
  g0[3] = (unsigned int)((ga >> 32) & 0x01FFFFFFu)   // global_addr[56:32]
          | (2u << 30);                              // type=2 ("image")

  i32x8 g1;
  g1[0] = (int)(2u << 16);              // data_size=2 -> 4-byte elements
  g1[1] = (int)((unsigned)(LK + 1) << 16); // tensor_dim0[15:0] @ bits[63:48]
  g1[2] = (int)(16u << 16);             // tensor_dim0[31:16]=0, tensor_dim1=16
  g1[3] = (int)((unsigned)TK << 16);    // tensor_dim1[31:16]=0, tile_dim0=64
  g1[4] = 16;                           // tile_dim1=16, tile_dim2=0
  g1[5] = LK + 1;                       // tensor_dim0_stride[31:0] = 2049
  g1[6] = 0;                            // stride hi, dim1_stride lo
  g1[7] = 0;

  i32x4 gz4 = {0, 0, 0, 0};             // groups 2/3 unused (2D tensor)
  i32x8 gz8 = {0, 0, 0, 0, 0, 0, 0, 0};
  __builtin_amdgcn_tensor_load_to_lds(g0, g1, gz4, gz4, gz8, 0);
}
#endif

__global__ __launch_bounds__(256)
void fa_sink_dropout_kernel(const float* __restrict__ qg,
                            const float* __restrict__ kg,
                            const float* __restrict__ vg,
                            const int*   __restrict__ maskg,
                            const float* __restrict__ ug,
                            float*       __restrict__ outg)
{
  constexpr float SCALE    = 0.08838834764831845f;  // 1/sqrt(128)
  constexpr float RATE     = 0.1f;
  constexpr float INV_KEEP = 1.0f / 0.9f;

  __shared__ __align__(32) __bf16 sK [TK * D_];        // [k][d], row major
  __shared__ __align__(32) __bf16 sVt[D_ * TK];        // [d][k], transposed V
  __shared__ __align__(32) __bf16 sP [NW * 16 * TK];   // per-wave P staging
  __shared__ float sMask[TK];
#if USE_TDM
  __shared__ __align__(16) float sU[2][NW * 16 * TK];  // double-buffered u tiles
#endif

  const int tid      = threadIdx.x;
  const int wave     = tid >> 5;
  const int lane     = tid & 31;
  const int laneN    = lane & 15;       // N / M-within-lane index
  const int laneHalf = lane >> 4;       // 0 or 1
  const int kb8      = laneHalf * 8;    // A-fragment K base
  const int kb16     = laneHalf * 16;   // unused (B fragments also use kb8)

  const int b  = blockIdx.y;
  const int q0 = blockIdx.x * TQ + wave * 16;   // this wave's first query row

  // ---- preload Q as WMMA A-fragments (16x32 bf16 per chunk, 4 chunks of D) ----
  v16bf aQ[4];
  {
    const float* qp = qg + (size_t)(b * LQ + q0 + laneN) * D_;
    #pragma unroll
    for (int c = 0; c < 4; ++c) {
      #pragma unroll
      for (int e = 0; e < 8; ++e) {
        aQ[c][e]     = (__bf16)qp[32 * c + kb8 + e];
        aQ[c][e + 8] = (__bf16)qp[32 * c + 16 + kb8 + e];
      }
    }
  }

  // Online-softmax state seeded with the attention sink (logit 0, V-row 0):
  // m = 0, l = 1 carries exp(0 - m) through every alpha rescale exactly.
  v8f O[8];
  float m_r[8], l_r[8];
  #pragma unroll
  for (int r = 0; r < 8; ++r) { O[r] = {}; m_r[r] = 0.0f; l_r[r] = 1.0f; }

  const size_t uBase = (size_t)(b * LQ) * (size_t)(LK + 1)
                     + (size_t)q0 * (size_t)(LK + 1);

#if USE_TDM
  // Prime the DMA pipeline: u tile for kt=0 into buffer 0.
  tdm_load_u_tile(ug + uBase, &sU[0][wave * 16 * TK]);
#endif

  #pragma unroll 1
  for (int kt = 0; kt < NT; ++kt) {
    const int kb  = kt * TK;
    const int buf = kt & 1;

    // ---- cooperative tile load: K (row major) and V (transposed) as bf16 ----
    for (int i = tid; i < TK * (D_ / 4); i += 256) {
      const int row = i >> 5;             // 32 float4 per 128-wide row
      const int c4  = (i & 31) << 2;
      const size_t g = (size_t)(b * LK + kb + row) * D_ + c4;
      const float4 kv = *(const float4*)(kg + g);
      const float4 vv = *(const float4*)(vg + g);
      if (kt + 1 < NT && i < 512) {       // warm L2 for next tile (prefetch_b8)
        __builtin_prefetch(kg + g + (size_t)TK * D_, 0, 1);
        __builtin_prefetch(vg + g + (size_t)TK * D_, 0, 1);
      }
      sK[row * D_ + c4 + 0] = (__bf16)kv.x;
      sK[row * D_ + c4 + 1] = (__bf16)kv.y;
      sK[row * D_ + c4 + 2] = (__bf16)kv.z;
      sK[row * D_ + c4 + 3] = (__bf16)kv.w;
      sVt[(c4 + 0) * TK + row] = (__bf16)vv.x;
      sVt[(c4 + 1) * TK + row] = (__bf16)vv.y;
      sVt[(c4 + 2) * TK + row] = (__bf16)vv.z;
      sVt[(c4 + 3) * TK + row] = (__bf16)vv.w;
    }
    if (tid < TK)
      sMask[tid] = -1.0e9f * (1.0f - (float)maskg[b * LK + kb + tid]);
    __syncthreads();

    // ---- S = scale*Q.K^T + mask : 4 n-tiles of 16x16 per wave ----
    v8f S[4];
    #pragma unroll
    for (int n = 0; n < 4; ++n) {
      v8f acc = {};
      #pragma unroll
      for (int c = 0; c < 4; ++c) {
        const __bf16* kp = &sK[(n * 16 + laneN) * D_ + 32 * c + kb8];
        const v16bf bk = __builtin_shufflevector(
            *(const v8bf*)(kp), *(const v8bf*)(kp + 16),
            0, 1, 2, 3, 4, 5, 6, 7, 8, 9, 10, 11, 12, 13, 14, 15);
        acc = __builtin_amdgcn_wmma_f32_16x16x32_bf16(
            false, aQ[c], false, bk, (short)0, acc, false, false);
      }
      const float madd = sMask[n * 16 + laneN];
      #pragma unroll
      for (int r = 0; r < 8; ++r) S[n][r] = S[n][r] * SCALE + madd;
    }

    // ---- online softmax update ----
    float alpha[8], mn[8];
    #pragma unroll
    for (int r = 0; r < 8; ++r) {
      float rmax = fmaxf(fmaxf(S[0][r], S[1][r]), fmaxf(S[2][r], S[3][r]));
      rmax = warp_max_half16(rmax);           // C layout: row spans a 16-lane half
      mn[r]    = fmaxf(m_r[r], rmax);
      alpha[r] = __expf(m_r[r] - mn[r]);
      m_r[r]   = mn[r];
    }
    #pragma unroll
    for (int no = 0; no < 8; ++no)
      #pragma unroll
      for (int r = 0; r < 8; ++r) O[no][r] *= alpha[r];

#if USE_TDM
    // Kick off DMA for the NEXT u tile, then wait for the CURRENT one.
    // TDM ops complete in order per wave: tensorcnt<=1 ==> tile kt landed.
    if (kt + 1 < NT) {
      tdm_load_u_tile(ug + uBase + (size_t)(kb + TK),
                      &sU[buf ^ 1][wave * 16 * TK]);
      __builtin_amdgcn_s_wait_tensorcnt(1);
    } else {
      __builtin_amdgcn_s_wait_tensorcnt(0);
    }
    const float* sUw = &sU[buf][wave * 16 * TK];
#endif

    // ---- P = exp(S-m); row sums BEFORE dropout; stage dropped P (bf16) ----
    float rsum[8];
    #pragma unroll
    for (int r = 0; r < 8; ++r) rsum[r] = 0.0f;
    #pragma unroll
    for (int n = 0; n < 4; ++n) {
      #pragma unroll
      for (int r = 0; r < 8; ++r) {
        const float p = __expf(S[n][r] - mn[r]);
        rsum[r] += p;                                   // denominator: pre-dropout
        const int M = r + kb8;                          // C layout row index
#if USE_TDM
        const float uu = sUw[M * TK + n * 16 + laneN];
#else
        const float uu =
            ug[uBase + (size_t)M * (LK + 1) + kb + n * 16 + laneN];
#endif
        const float pd = (uu >= RATE) ? p : 0.0f;       // inverted dropout (1/0.9 at end)
        sP[wave * (16 * TK) + M * TK + n * 16 + laneN] = (__bf16)pd;
      }
    }
    #pragma unroll
    for (int r = 0; r < 8; ++r)
      l_r[r] = l_r[r] * alpha[r] + warp_sum_half16(rsum[r]);

    // ---- O += P.V : read P back in A layout, V B-fragments from sVt ----
    #pragma unroll
    for (int c = 0; c < 2; ++c) {
      const __bf16* pw = &sP[wave * (16 * TK) + laneN * TK + 32 * c + kb8];
      const v8bf lo = *(const v8bf*)(pw);
      const v8bf hi = *(const v8bf*)(pw + 16);
      const v16bf aP = __builtin_shufflevector(
          lo, hi, 0, 1, 2, 3, 4, 5, 6, 7, 8, 9, 10, 11, 12, 13, 14, 15);
      #pragma unroll
      for (int no = 0; no < 8; ++no) {
        const __bf16* vp = &sVt[(no * 16 + laneN) * TK + 32 * c + kb8];
        const v16bf bv = __builtin_shufflevector(
            *(const v8bf*)(vp), *(const v8bf*)(vp + 16),
            0, 1, 2, 3, 4, 5, 6, 7, 8, 9, 10, 11, 12, 13, 14, 15);
        O[no] = __builtin_amdgcn_wmma_f32_16x16x32_bf16(
            false, aP, false, bv, (short)0, O[no], false, false);
      }
    }
    __syncthreads();   // before next tile overwrites sK/sVt
  }

  // ---- finalize: Z already includes the sink term via the l=1 seed ----
  #pragma unroll
  for (int r = 0; r < 8; ++r) l_r[r] = INV_KEEP / l_r[r];

  float* op = outg + (size_t)(b * LQ) * D_;
  #pragma unroll
  for (int no = 0; no < 8; ++no) {
    #pragma unroll
    for (int r = 0; r < 8; ++r) {
      const int M = r + kb8;
      op[(size_t)(q0 + M) * D_ + no * 16 + laneN] = O[no][r] * l_r[r];
    }
  }
}

extern "C" void kernel_launch(void* const* d_in, const int* in_sizes, int n_in,
                              void* d_out, int out_size, void* d_ws, size_t ws_size,
                              hipStream_t stream) {
  (void)in_sizes; (void)n_in; (void)out_size; (void)d_ws; (void)ws_size;
  const float* q    = (const float*)d_in[0];
  const float* k    = (const float*)d_in[1];
  const float* v    = (const float*)d_in[2];
  const int*   mask = (const int*)d_in[3];
  const float* u    = (const float*)d_in[4];
  float* out        = (float*)d_out;

  dim3 grid(LQ / TQ, B_);   // 16 x 16 blocks
  fa_sink_dropout_kernel<<<grid, 256, 0, stream>>>(q, k, v, mask, u, out);
}
